// OnlyAttentionFixErrorModelWithoutInputEmbedding_2422361555029
// MI455X (gfx1250) — matrix operand order repaired and checked by reading, not compiled, not run on hardware
//
#include <hip/hip_runtime.h>

// ---------------------------------------------------------------------------
// Problem constants (from the reference)
// ---------------------------------------------------------------------------
constexpr int kB     = 4;
constexpr int kS     = 1024;
constexpr int kH     = 512;
constexpr int kHeads = 8;
constexpr int kDH    = 64;        // kH / kHeads
constexpr int kV     = 32000;
constexpr int kTok   = kB * kS;   // 4096 rows in every big GEMM

using bf16 = __bf16;

typedef __attribute__((ext_vector_type(16))) bf16  v16bf;
typedef __attribute__((ext_vector_type(8)))  float v8f;
typedef unsigned int u32x4 __attribute__((ext_vector_type(4)));
typedef int          i32x4 __attribute__((ext_vector_type(4)));
typedef int          i32x8 __attribute__((ext_vector_type(8)));

union frag16 {            // one 16x32 (A) or 32x16 (B) bf16 WMMA fragment
  v16bf v;
  uint4 u[2];
};

#ifndef USE_TDM
#define USE_TDM 1
#endif

// ---------------------------------------------------------------------------
// Positional encoding:  y = bf16( x + [sin(s*inv_c) | cos(s*inv_c)] )
// ---------------------------------------------------------------------------
__global__ __launch_bounds__(256) void posenc_kernel(
    const float* __restrict__ x, bf16* __restrict__ y, int total)
{
  int i = blockIdx.x * blockDim.x + threadIdx.x;
  if (i >= total) return;
  int c = i % kH;
  int s = (i / kH) % kS;
  const int n = kH / 2;
  int cc = (c < n) ? c : c - n;
  float freq = __expf((float)cc * (-__logf(10000.0f) / (float)(n - 1)));
  float ang  = (float)s * freq;
  float sig  = (c < n) ? __sinf(ang) : __cosf(ang);
  y[i] = (bf16)(x[i] + sig);
}

// ---------------------------------------------------------------------------
// Generic WMMA GEMM:  C[M,N] = A[M,K] (bf16) @ W[K,N] (f32->bf16) + bias
// Block: 256 threads = 8 waves; C tile 128(M) x 64(N); waves 4x2 of 32x32.
// A tile staged by the Tensor Data Mover (TDM): 2D descriptor, 128x32 bf16
// tile with LDS padding 4 DWORDs every 16 DWORDs -> 40-element rows.
// flags: bit0 = ReLU, bit1 = f32 output (else bf16 output).
// ---------------------------------------------------------------------------
__global__ __launch_bounds__(256) void gemm_bf16_kernel(
    const bf16* __restrict__ A, int lda,
    const float* __restrict__ W, int ldw,
    const float* __restrict__ bias,
    void* __restrict__ C, int ldc,
    int M, int N, int K, int flags)
{
  __shared__ __align__(16) bf16 lA[128][40];   // 128 x 32, padded (80B rows)
  __shared__ __align__(16) bf16 lB[32][72];    // 32 x 64,  padded (144B rows)

  const int tid   = threadIdx.x;
  const int lane  = tid & 31;
  const int wid   = tid >> 5;     // 0..7
  const int wm    = wid >> 1;     // 0..3  (M direction)
  const int wn    = wid & 1;      // 0..1  (N direction)
  const int lidx  = lane & 15;
  const int khalf = lane >> 4;
  const int m0    = blockIdx.y * 128;
  const int n0    = blockIdx.x * 64;

  v8f acc[2][2] = {{{}, {}}, {{}, {}}};

  for (int k0 = 0; k0 < K; k0 += 32) {
    // ---- stage A tile via TDM (wave 0 issues one DMA per K-step) ----
#if USE_TDM
    if (wid == 0) {
      unsigned long long ga =
          (unsigned long long)(uintptr_t)(A + (size_t)m0 * lda + k0);
      unsigned int ldsA = (unsigned int)(uintptr_t)&lA[0][0];
      // D# group 0: count=1 | lds_addr | global_addr[56:0] | type=2
      u32x4 g0 = { 1u,
                   ldsA,
                   (unsigned int)(ga & 0xffffffffull),
                   (unsigned int)((ga >> 32) & 0x1ffffffull) | (2u << 30) };
      // D# group 1: data_size=2B, pad_enable, pad_interval=16 DW (code 3),
      // pad_amount=4 DW (code 3); tensor_dim0=K, tensor_dim1=M,
      // tile_dim0=32, tile_dim1=128, tensor_dim0_stride=lda.
      i32x8 g1 = { (int)((1u << 16) | (1u << 20) | (3u << 22) | (3u << 25)),
                   (int)(((unsigned int)K & 0xffffu) << 16),
                   (int)((((unsigned int)K >> 16) & 0xffffu) |
                         (((unsigned int)M & 0xffffu) << 16)),
                   (int)((((unsigned int)M >> 16) & 0xffffu) | (32u << 16)),
                   (int)128,                 // tile_dim1=128, tile_dim2=0
                   (int)(unsigned int)lda,   // dim0_stride[31:0]
                   0, 0 };                   // dim0_stride[47:32]=0, dim1_stride=0
      i32x4 gz = { 0, 0, 0, 0 };
#if defined(__clang_major__) && (__clang_major__ >= 23)
      i32x8 gz8 = { 0, 0, 0, 0, 0, 0, 0, 0 };
      __builtin_amdgcn_tensor_load_to_lds(g0, g1, gz, gz, gz8, 0);
#else
      __builtin_amdgcn_tensor_load_to_lds(g0, g1, gz, gz, 0);
#endif
      __builtin_amdgcn_s_wait_tensorcnt(0);
    }
#else
    {
      int r = tid >> 2;
      int c = (tid & 3) * 8;
      for (int p = 0; p < 2; ++p, r += 64) {
        int gr = m0 + r;
        uint4 val = make_uint4(0u, 0u, 0u, 0u);
        if (gr < M)
          val = *reinterpret_cast<const uint4*>(A + (size_t)gr * lda + k0 + c);
        *reinterpret_cast<uint4*>(&lA[r][c]) = val;
      }
    }
#endif
    // ---- stage W tile: 32x64 f32 -> bf16, row-major (B wants lane = n) ----
    {
      int kr = tid >> 4;
      int nc = (tid & 15) * 4;
      for (int p = 0; p < 2; ++p, kr += 16) {
        float4 wv = make_float4(0.f, 0.f, 0.f, 0.f);
        if (n0 + nc + 3 < N) {
          wv = *reinterpret_cast<const float4*>(W + (size_t)(k0 + kr) * ldw + n0 + nc);
          if (k0 + 32 < K)   // prefetch next W tile (global_prefetch_b8)
            __builtin_prefetch(W + (size_t)(k0 + 32 + kr) * ldw + n0 + nc, 0, 1);
        }
        lB[kr][nc + 0] = (bf16)wv.x;
        lB[kr][nc + 1] = (bf16)wv.y;
        lB[kr][nc + 2] = (bf16)wv.z;
        lB[kr][nc + 3] = (bf16)wv.w;
      }
    }
    __syncthreads();

    // ---- fragments + 4 WMMAs per wave ----
    frag16 af[2], bfr[2];
    for (int i = 0; i < 2; ++i) {
      int row = wm * 32 + i * 16 + lidx;                     // A: lane = M
      af[i].u[0] = *reinterpret_cast<const uint4*>(&lA[row][khalf * 8]);
      af[i].u[1] = *reinterpret_cast<const uint4*>(&lA[row][16 + khalf * 8]);
    }
    for (int i = 0; i < 2; ++i) {
      int col = wn * 32 + i * 16 + lidx;                     // B: lane = N
      for (int s = 0; s < 8; ++s) {
        bfr[i].v[s]     = lB[khalf * 8 + s][col];
        bfr[i].v[8 + s] = lB[16 + khalf * 8 + s][col];
      }
    }
    for (int mi = 0; mi < 2; ++mi)
      for (int ni = 0; ni < 2; ++ni)
        acc[mi][ni] = __builtin_amdgcn_wmma_f32_16x16x32_bf16(
            false, af[mi].v, false, bfr[ni].v, (short)0, acc[mi][ni], false, false);
    __syncthreads();
  }

  // ---- epilogue: bias, ReLU, store (C layout: M = khalf*8 + v, N = lidx) ----
  const bool relu   = (flags & 1) != 0;
  const bool f32out = (flags & 2) != 0;
  for (int mi = 0; mi < 2; ++mi)
    for (int ni = 0; ni < 2; ++ni) {
      int rbase = m0 + wm * 32 + mi * 16 + khalf * 8;
      int col   = n0 + wn * 32 + ni * 16 + lidx;
      if (col >= N) continue;
      float bv = bias ? bias[col] : 0.0f;
      for (int v = 0; v < 8; ++v) {
        int r = rbase + v;
        if (r >= M) continue;
        float val = acc[mi][ni][v] + bv;
        if (relu) val = fmaxf(val, 0.0f);
        if (f32out) ((float*)C)[(size_t)r * ldc + col] = val;
        else        ((bf16*)C)[(size_t)r * ldc + col] = (bf16)val;
      }
    }
}

// ---------------------------------------------------------------------------
// Flash-style attention: softmax(Q K^T / 8, key-mask) V, per (b, head).
// Block = 128 threads (4 waves), each wave owns a 16-query tile; key blocks
// of 32 with online softmax (row stats live in the C-layout halves, reduced
// with __shfl_xor over the 16-lane group).  Q/K/V are [B,S,H] bf16.
// ---------------------------------------------------------------------------
__global__ __launch_bounds__(128) void attn_kernel(
    const bf16* __restrict__ Q, const bf16* __restrict__ K,
    const bf16* __restrict__ V, const unsigned char* __restrict__ mask,
    bf16* __restrict__ O)
{
  __shared__ __align__(16) bf16 ldsKT[kDH][40];     // K-block transposed [d][key]
  __shared__ __align__(16) bf16 ldsP[4][16][40];    // per-wave probs 16x32

  const int tid   = threadIdx.x;
  const int lane  = tid & 31;
  const int wv    = tid >> 5;
  const int lidx  = lane & 15;
  const int khalf = lane >> 4;
  const int b     = blockIdx.z;
  const int h     = blockIdx.y;
  const int q0    = blockIdx.x * 64 + wv * 16;

  const bf16* Qh = Q + (size_t)b * kS * kH + h * kDH;
  const bf16* Kh = K + (size_t)b * kS * kH + h * kDH;
  const bf16* Vh = V + (size_t)b * kS * kH + h * kDH;

  // Q fragments (A layout: lane = query row), j=0 -> d 0..31, j=1 -> d 32..63
  frag16 qf[2];
  for (int j = 0; j < 2; ++j) {
    const bf16* qp = Qh + (size_t)(q0 + lidx) * kH + j * 32 + khalf * 8;
    qf[j].u[0] = *reinterpret_cast<const uint4*>(qp);
    qf[j].u[1] = *reinterpret_cast<const uint4*>(qp + 16);
  }

  float rmax[8], rsum[8];
  v8f oacc[4] = {{}, {}, {}, {}};
  for (int v = 0; v < 8; ++v) { rmax[v] = -__builtin_inff(); rsum[v] = 0.0f; }

  for (int kb = 0; kb < kS; kb += 32) {
    __syncthreads();
    // stage K^T block: 32 keys x 64 d  ->  ldsKT[d][key]
    {
      int key = tid >> 2;
      int d0  = (tid & 3) * 16;
      const bf16* kp = Kh + (size_t)(kb + key) * kH + d0;
      for (int c = 0; c < 16; ++c) ldsKT[d0 + c][key] = kp[c];
    }
    __syncthreads();

    // scores: two 16x16 tiles (keys kb..+15, kb+16..+31)
    v8f sacc[2] = {{}, {}};
    for (int nt = 0; nt < 2; ++nt)
      for (int j = 0; j < 2; ++j) {
        frag16 kf;                                    // B layout: lane = key (N dim)
        for (int s = 0; s < 8; ++s) {
          kf.v[s]     = ldsKT[j * 32 + khalf * 8 + s][nt * 16 + lidx];
          kf.v[8 + s] = ldsKT[j * 32 + 16 + khalf * 8 + s][nt * 16 + lidx];
        }
        sacc[nt] = __builtin_amdgcn_wmma_f32_16x16x32_bf16(
            false, qf[j].v, false, kf.v, (short)0, sacc[nt], false, false);
      }

    // scale 1/sqrt(dh) and key mask (col = kb + nt*16 + lidx)
    for (int nt = 0; nt < 2; ++nt) {
      bool mk = mask[(size_t)b * kS + kb + nt * 16 + lidx] != 0;
      for (int v = 0; v < 8; ++v)
        sacc[nt][v] = mk ? sacc[nt][v] * 0.125f : -__builtin_inff();
    }

    // online softmax: per-row (row = khalf*8 + v) max across the 16 lanes
    float bm[8];
    for (int v = 0; v < 8; ++v) bm[v] = fmaxf(sacc[0][v], sacc[1][v]);
    for (int off = 8; off >= 1; off >>= 1)
      for (int v = 0; v < 8; ++v) bm[v] = fmaxf(bm[v], __shfl_xor(bm[v], off, 32));

    float ps[8];
    for (int v = 0; v < 8; ++v) {
      float nm = fmaxf(rmax[v], bm[v]);
      float f  = __expf(rmax[v] - nm);
      rmax[v]  = nm;
      rsum[v] *= f;
      for (int t = 0; t < 4; ++t) oacc[t][v] *= f;
      float p0 = __expf(sacc[0][v] - nm);
      float p1 = __expf(sacc[1][v] - nm);
      ps[v] = p0 + p1;
      int r = khalf * 8 + v;
      ldsP[wv][r][lidx]      = (bf16)p0;              // C layout -> LDS (row-major)
      ldsP[wv][r][16 + lidx] = (bf16)p1;
    }
    for (int off = 8; off >= 1; off >>= 1)
      for (int v = 0; v < 8; ++v) ps[v] += __shfl_xor(ps[v], off, 32);
    for (int v = 0; v < 8; ++v) rsum[v] += ps[v];

    // reload P as an A fragment (same-wave LDS ordering)
    frag16 pf;
    const bf16* pp = &ldsP[wv][lidx][khalf * 8];
    pf.u[0] = *reinterpret_cast<const uint4*>(pp);
    pf.u[1] = *reinterpret_cast<const uint4*>(pp + 16);

    // P @ V : B fragment straight from global V (lane = d, regs = 16 keys)
    for (int dt = 0; dt < 4; ++dt) {
      frag16 vf;
      for (int s = 0; s < 8; ++s) {
        vf.v[s]     = Vh[(size_t)(kb + khalf * 8 + s) * kH + dt * 16 + lidx];
        vf.v[8 + s] = Vh[(size_t)(kb + 16 + khalf * 8 + s) * kH + dt * 16 + lidx];
      }
      oacc[dt] = __builtin_amdgcn_wmma_f32_16x16x32_bf16(
          false, pf.v, false, vf.v, (short)0, oacc[dt], false, false);
    }
  }

  // epilogue: normalize and store [B,S,H]-layout head slice
  for (int dt = 0; dt < 4; ++dt)
    for (int v = 0; v < 8; ++v) {
      int r = khalf * 8 + v;
      int d = dt * 16 + lidx;
      O[((size_t)b * kS + q0 + r) * kH + h * kDH + d] = (bf16)(oacc[dt][v] / rsum[v]);
    }
}

// ---------------------------------------------------------------------------
// Host orchestration.  Param leaves follow jax pytree (sorted-key) order:
//   [0]=input [1]=input_mask [2]=output [3]=output_mask, then params:
//   dec[l] (24 leaves: ca{k,o,q,t,v}{b,w}, ff{f1,f2}{b,w}, sa{...}),
//   enc[l] (14 leaves: ff, sa), out{b,w}.  MHA offsets: kb,kw,ob,ow,qb,qw,tb,tw,vb,vw.
// ---------------------------------------------------------------------------
extern "C" void kernel_launch(void* const* d_in, const int* in_sizes, int n_in,
                              void* d_out, int out_size, void* d_ws, size_t ws_size,
                              hipStream_t stream)
{
  (void)in_sizes; (void)n_in; (void)out_size; (void)ws_size;

  const float*         x_in  = (const float*)d_in[0];
  const unsigned char* imask = (const unsigned char*)d_in[1];
  const float*         y_in  = (const float*)d_in[2];
  const unsigned char* omask = (const unsigned char*)d_in[3];
  auto F = [&](int i) -> const float* { return (const float*)d_in[i]; };

  const size_t bufE = (size_t)kTok * kH;          // 2 Mi bf16 elems per buffer
  bf16* base = (bf16*)d_ws;
  bf16* encX = base + 0 * bufE;
  bf16* decX = base + 1 * bufE;
  bf16* mo   = base + 2 * bufE;
  bf16* qbuf = base + 3 * bufE;
  bf16* kbuf = base + 4 * bufE;
  bf16* vbuf = base + 5 * bufE;
  bf16* attb = base + 6 * bufE;
  bf16* tmpb = base + 7 * bufE;

  auto gemm = [&](const bf16* A, int lda, const float* W, int ldw, const float* bias,
                  void* C, int ldc, int M, int N, int K, int flags) {
    dim3 grid((unsigned)((N + 63) / 64), (unsigned)((M + 127) / 128));
    gemm_bf16_kernel<<<grid, 256, 0, stream>>>(A, lda, W, ldw, bias, C, ldc, M, N, K, flags);
  };
  auto mha = [&](const bf16* x, const bf16* mem, const unsigned char* msk, int p, bf16* out) {
    gemm(x,   kH, F(p + 5), kH, F(p + 4), qbuf, kH, kTok, kH, kH, 0);   // q
    gemm(mem, kH, F(p + 1), kH, F(p + 0), kbuf, kH, kTok, kH, kH, 0);   // k
    gemm(mem, kH, F(p + 9), kH, F(p + 8), vbuf, kH, kTok, kH, kH, 0);   // v
    attn_kernel<<<dim3(kS / 64, kHeads, kB), 128, 0, stream>>>(qbuf, kbuf, vbuf, msk, attb);
    for (int hh = 0; hh < kHeads; ++hh)                                  // shared t (dh x dh)
      gemm(attb + hh * kDH, kH, F(p + 7), kDH, F(p + 6), tmpb + hh * kDH, kH, kTok, kDH, kDH, 0);
    gemm(tmpb, kH, F(p + 3), kH, F(p + 2), out, kH, kTok, kH, kH, 0);   // o
  };
  auto ff = [&](const bf16* x, int p, bf16* out) {
    gemm(x,    kH, F(p + 1), kH, F(p + 0), qbuf, kH, kTok, kH, kH, 1);  // f1 + ReLU
    gemm(qbuf, kH, F(p + 3), kH, F(p + 2), out,  kH, kTok, kH, kH, 0);  // f2
  };

  posenc_kernel<<<(kTok * kH) / 256, 256, 0, stream>>>(x_in, encX, kTok * kH);
  posenc_kernel<<<(kTok * kH) / 256, 256, 0, stream>>>(y_in, decX, kTok * kH);

  const int P = 4;
  for (int l = 0; l < 2; ++l) {                   // encoder
    int eb = P + 48 + 14 * l;                     // ff@+0, sa@+4
    mha(encX, encX, imask, eb + 4, mo);
    ff(mo, eb + 0, encX);
  }
  for (int l = 0; l < 2; ++l) {                   // decoder
    int db = P + 24 * l;                          // ca@+0, ff@+10, sa@+14
    mha(decX, decX, omask, db + 14, mo);          // masked self-attention
    mha(mo, encX, imask, db + 0, decX);           // cross-attention
    ff(decX, db + 10, decX);
  }
  // vocab logits, f32 straight into d_out (134 GF — the dominant GEMM)
  gemm(decX, kH, F(P + 77), kV, F(P + 76), d_out, kV, kTok, kV, kH, 2);
}
